// Transformer_54992761258385
// MI455X (gfx1250) — matrix operand order repaired and checked by reading, not compiled, not run on hardware
//
#include <hip/hip_runtime.h>
#include <math.h>

// ---------------------------------------------------------------------------
// Transformer block for MI455X (gfx1250): bf16 WMMA + async-to-LDS staging.
//   dims: x[4,2048,1024], heads=16, dhead=64, mlp=4096
// ---------------------------------------------------------------------------

typedef __bf16 bf16_t;
typedef __attribute__((ext_vector_type(16))) __bf16     v16bf;
typedef __attribute__((ext_vector_type(8)))  __bf16     v8bf;
typedef __attribute__((ext_vector_type(8)))  float      v8f;
typedef __attribute__((ext_vector_type(8)))  unsigned   v8u;

#define DIMX  1024
#define SEQ   2048
#define BATCH 4
#define HEADS 16
#define DHEAD 64
#define MLPD  4096
#define ROWS  (BATCH * SEQ) /* 8192 */

// ---------------- helpers ----------------

__device__ inline v8f v8f_zero() {
  v8f z = {0.f, 0.f, 0.f, 0.f, 0.f, 0.f, 0.f, 0.f};
  return z;
}

__device__ inline v8f wmma_bf16f32(v16bf a, v16bf b, v8f c) {
  // D = A x B + C  (16x16x32, bf16 in, f32 accumulate)
  return __builtin_amdgcn_wmma_f32_16x16x32_bf16(
      /*neg_a=*/false, a, /*neg_b=*/false, b,
      /*c_mod=*/(short)0, c, /*reuse_a=*/false, /*reuse_b=*/false);
}

// A-matrix fragment (16x32 bf16, MxK). Contiguous k order used here: lane<16 holds K 0..15,
// lane>=16 holds K 16..31; m = lane&15. Two contiguous 16B loads
// (global_load_b128 or ds_load_b128 depending on the pointer).
__device__ inline v16bf load_fragA(const bf16_t* base, int mBase, int ld,
                                   int kBase, int lane) {
  const int m    = mBase + (lane & 15);
  const int koff = (lane >> 4) << 4; // 0 or 16
  const bf16_t* p = base + (size_t)m * ld + kBase + koff;
  v8bf lo = *(const v8bf*)p;        // K = kBase+koff    .. +7
  v8bf hi = *(const v8bf*)(p + 8);  // K = kBase+koff+8  .. +15
  return __builtin_shufflevector(lo, hi, 0, 1, 2, 3, 4, 5, 6, 7,
                                 8, 9, 10, 11, 12, 13, 14, 15);
}

// B-matrix fragment (32x16 bf16, KxN). Contiguous k order used here: lane<16 holds K 0..15,
// lane>=16 holds K 16..31; n = lane&15. One contiguous 32B load. Source must be
// N-major / K-contiguous.
__device__ inline v16bf load_fragB(const bf16_t* base, int nBase, int ld,
                                   int kBase, int lane) {
  const int n = nBase + (lane & 15);
  const int k = kBase + ((lane >> 4) << 4); // 0 or 16
  return *(const v16bf*)(base + (size_t)n * ld + k);
}

// lane <-> lane^mask exchange (group-of-32 ds_swizzle, wave32)
template <int PAT>
__device__ inline float dswz(float x) {
  return __int_as_float(__builtin_amdgcn_ds_swizzle(__float_as_int(x), PAT));
}
template <int PAT>
__device__ inline unsigned dswzu(unsigned x) {
  return (unsigned)__builtin_amdgcn_ds_swizzle((int)x, PAT);
}
#define SWAPX16 0x401F
#define SWAPX8  0x201F
#define SWAPX4  0x101F
#define SWAPX2  0x081F
#define SWAPX1  0x041F

__device__ inline unsigned pack2bf(float a, float b) {
  union { __bf16 h[2]; unsigned u; } t;
  t.h[0] = (__bf16)a;
  t.h[1] = (__bf16)b;
  return t.u;
}

// CDNA5 async copy: global -> LDS, 16B per lane, tracked by ASYNCcnt.
__device__ inline void async_copy_b128(unsigned lds_off, const bf16_t* src) {
  asm volatile("global_load_async_to_lds_b128 %0, %1, off"
               :: "v"(lds_off), "v"(src)
               : "memory");
}
__device__ inline unsigned lds_off32(const void* p) {
  return (unsigned)(size_t)p; // generic LDS addr: low 32 bits = LDS offset
}

// ---------------- GEMM core: 128(M) x 128(N) block tile, 8 waves ----------
// Per wave: 32x64 output tile (8 v8f accumulators). A and B panels
// (128 x 64 per stage, 16KB each) async-staged into LDS, triple buffered
// via rotating pointers (no %3 math); a buffer is rewritten only one
// barrier after its last read. Each wave issues 8 async ops per stage, so
// s_wait_asynccnt 0x8 guarantees the previous stage has landed.
// The stage loop is NOT unrolled so accumulators stay pinned in one
// register tuple (no v_mov/v_nop hazard churn around v_wmma).

__device__ inline void stage_panel64(const bf16_t* src, int ldSrc, int rowBase,
                                     int kb, bf16_t* dst) {
  // 128 rows x 64 bf16 = 16KB = 1024 x 16B chunks; 256 threads x 4 chunks
  const int t = threadIdx.x;
#pragma unroll
  for (int j = 0; j < 4; ++j) {
    const int chunk = t + 256 * j;
    const int row = chunk >> 3, ch = chunk & 7;
    async_copy_b128(lds_off32(dst + row * 64 + ch * 8),
                    src + (size_t)(rowBase + row) * ldSrc + kb + ch * 8);
  }
}

#define PSTAGE (128 * 64) /* elements per stage buffer */

__device__ inline void gemm_block_128x128(const bf16_t* __restrict__ A,
                                          const bf16_t* __restrict__ Bt, int K,
                                          int mBlock, int nBlock,
                                          bf16_t* sA /* [3][128*64] */,
                                          bf16_t* sB /* [3][128*64] */,
                                          v8f acc[2][4]) {
  const int lane = threadIdx.x & 31;
  const int w    = threadIdx.x >> 5;
  const int mSub = (w >> 1) * 32; // 4 M-groups
  const int nSub = (w & 1) * 64;  // 2 N-groups
#pragma unroll
  for (int mt = 0; mt < 2; ++mt)
#pragma unroll
    for (int nt = 0; nt < 4; ++nt) acc[mt][nt] = v8f_zero();

  const int S = K / 64;
  bf16_t* rdA = sA;              bf16_t* rdB = sB;
  bf16_t* wrA = sA + PSTAGE;     bf16_t* wrB = sB + PSTAGE;
  bf16_t* spA = sA + 2 * PSTAGE; bf16_t* spB = sB + 2 * PSTAGE;
  stage_panel64(A, K, mBlock, 0, rdA);
  stage_panel64(Bt, K, nBlock, 0, rdB);
#pragma clang loop unroll(disable)
  for (int i = 0; i < S; ++i) {
    if (i + 1 < S) {
      stage_panel64(A, K, mBlock, (i + 1) * 64, wrA);
      stage_panel64(Bt, K, nBlock, (i + 1) * 64, wrB);
      asm volatile("s_wait_asynccnt 0x8" ::: "memory"); // stage i landed
    } else {
      asm volatile("s_wait_asynccnt 0x0" ::: "memory");
    }
    __syncthreads();
#pragma unroll
    for (int kk = 0; kk < 64; kk += 32) {
      const v16bf af0 = load_fragA(rdA, mSub, 64, kk, lane);
      const v16bf af1 = load_fragA(rdA, mSub + 16, 64, kk, lane);
#pragma unroll
      for (int nt = 0; nt < 4; ++nt) {
        const v16bf bfrag = load_fragB(rdB, nSub + 16 * nt, 64, kk, lane);
        acc[0][nt] = wmma_bf16f32(af0, bfrag, acc[0][nt]);
        acc[1][nt] = wmma_bf16f32(af1, bfrag, acc[1][nt]);
      }
    }
    bf16_t* tA = rdA; rdA = wrA; wrA = spA; spA = tA; // rotate triple buffer
    bf16_t* tB = rdB; rdB = wrB; wrB = spB; spB = tB;
  }
}

// ---------------- prep kernels ----------------

// out[n*K + k] = (bf16) in[k*N + n]   (in is [K][N] fp32)
__global__ __launch_bounds__(256) void k_transpose_cvt(
    const float* __restrict__ in, bf16_t* __restrict__ out, int K, int N) {
  size_t idx = (size_t)blockIdx.x * 256 + threadIdx.x;
  if (idx >= (size_t)K * N) return;
  int k = (int)(idx / N);
  int n = (int)(idx % N);
  out[(size_t)n * K + k] = (bf16_t)in[idx];
}

// row layernorm (1024 cols) -> bf16
__global__ __launch_bounds__(256) void k_layernorm(const float* __restrict__ x,
                                                   const float* __restrict__ g,
                                                   const float* __restrict__ bb,
                                                   bf16_t* __restrict__ out) {
  const int row = blockIdx.x;
  const float* xr = x + (size_t)row * DIMX;
  float v[4], s = 0.f, s2 = 0.f;
#pragma unroll
  for (int i = 0; i < 4; ++i) {
    v[i] = xr[threadIdx.x + 256 * i];
    s += v[i];
    s2 += v[i] * v[i];
  }
  s += dswz<SWAPX16>(s);  s2 += dswz<SWAPX16>(s2);
  s += dswz<SWAPX8>(s);   s2 += dswz<SWAPX8>(s2);
  s += dswz<SWAPX4>(s);   s2 += dswz<SWAPX4>(s2);
  s += dswz<SWAPX2>(s);   s2 += dswz<SWAPX2>(s2);
  s += dswz<SWAPX1>(s);   s2 += dswz<SWAPX1>(s2);
  __shared__ float red0[8], red1[8];
  if ((threadIdx.x & 31) == 0) {
    red0[threadIdx.x >> 5] = s;
    red1[threadIdx.x >> 5] = s2;
  }
  __syncthreads();
  float ts = 0.f, ts2 = 0.f;
#pragma unroll
  for (int i = 0; i < 8; ++i) { ts += red0[i]; ts2 += red1[i]; }
  const float mu  = ts * (1.f / DIMX);
  const float var = ts2 * (1.f / DIMX) - mu * mu;
  const float inv = rsqrtf(var + 1e-5f);
  bf16_t* orow = out + (size_t)row * DIMX;
#pragma unroll
  for (int i = 0; i < 4; ++i) {
    int c = threadIdx.x + 256 * i;
    orow[c] = (bf16_t)((v[i] - mu) * inv * g[c] + bb[c]);
  }
}

// ---------------- GEMM kernels ----------------

// QKV projection: [8192,1024] x [1024,3072]; epilogue scatters q,k as
// [b,h,n,d] bf16 and v transposed as [b,h,d,n] bf16.
__global__ __launch_bounds__(256) void k_gemm_qkv(
    const bf16_t* __restrict__ A, const bf16_t* __restrict__ Bt,
    bf16_t* __restrict__ qb, bf16_t* __restrict__ kb,
    bf16_t* __restrict__ vtb) {
  __shared__ __align__(16) bf16_t sA[3 * PSTAGE];
  __shared__ __align__(16) bf16_t sB[3 * PSTAGE];
  const int lane = threadIdx.x & 31;
  const int w    = threadIdx.x >> 5;
  const int WNB = (3 * DIMX) / 128; // 24
  const int mBlock = (blockIdx.x / WNB) * 128;
  const int nBlock = (blockIdx.x % WNB) * 128;
  v8f acc[2][4];
  gemm_block_128x128(A, Bt, DIMX, mBlock, nBlock, sA, sB, acc);
  const int mBaseW = mBlock + (w >> 1) * 32;
  const int nBaseW = nBlock + (w & 1) * 64;
  const int mo = (lane >> 4) << 3;
  const int no = lane & 15;
#pragma unroll
  for (int mt = 0; mt < 2; ++mt)
#pragma unroll
    for (int nt = 0; nt < 4; ++nt)
#pragma unroll
      for (int r = 0; r < 8; ++r) {
        const int row = mBaseW + 16 * mt + mo + r;
        const int col = nBaseW + 16 * nt + no;
        const int b = row >> 11, n = row & (SEQ - 1);
        const float val = acc[mt][nt][r];
        if (col < DIMX) {
          const int h = col >> 6, d = col & 63;
          qb[((size_t)(b * HEADS + h) * SEQ + n) * DHEAD + d] = (bf16_t)val;
        } else if (col < 2 * DIMX) {
          const int c = col - DIMX, h = c >> 6, d = c & 63;
          kb[((size_t)(b * HEADS + h) * SEQ + n) * DHEAD + d] = (bf16_t)val;
        } else {
          const int c = col - 2 * DIMX, h = c >> 6, d = c & 63;
          vtb[((size_t)(b * HEADS + h) * DHEAD + d) * SEQ + n] = (bf16_t)val;
        }
      }
}

// generic: out_f32[row,col] = acc + bias[col] + resid[row,col]
__global__ __launch_bounds__(256) void k_gemm_bias_res(
    const bf16_t* __restrict__ A, const bf16_t* __restrict__ Bt,
    const float* __restrict__ bias, const float* __restrict__ resid,
    float* __restrict__ outp, int K, int N) {
  __shared__ __align__(16) bf16_t sA[3 * PSTAGE];
  __shared__ __align__(16) bf16_t sB[3 * PSTAGE];
  const int lane = threadIdx.x & 31;
  const int w    = threadIdx.x >> 5;
  const int WNB = N / 128;
  const int mBlock = (blockIdx.x / WNB) * 128;
  const int nBlock = (blockIdx.x % WNB) * 128;
  v8f acc[2][4];
  gemm_block_128x128(A, Bt, K, mBlock, nBlock, sA, sB, acc);
  const int mBaseW = mBlock + (w >> 1) * 32;
  const int nBaseW = nBlock + (w & 1) * 64;
  const int mo = (lane >> 4) << 3;
  const int no = lane & 15;
#pragma unroll
  for (int mt = 0; mt < 2; ++mt)
#pragma unroll
    for (int nt = 0; nt < 4; ++nt)
#pragma unroll
      for (int r = 0; r < 8; ++r) {
        const int row = mBaseW + 16 * mt + mo + r;
        const int col = nBaseW + 16 * nt + no;
        const size_t idx = (size_t)row * N + col;
        outp[idx] = acc[mt][nt][r] + bias[col] + resid[idx];
      }
}

// MLP up-proj with exact GELU, bf16 output. [8192,1024] x [1024,4096]
__global__ __launch_bounds__(256) void k_gemm_mlp1(
    const bf16_t* __restrict__ A, const bf16_t* __restrict__ Bt,
    const float* __restrict__ bias, bf16_t* __restrict__ outp) {
  __shared__ __align__(16) bf16_t sA[3 * PSTAGE];
  __shared__ __align__(16) bf16_t sB[3 * PSTAGE];
  const int lane = threadIdx.x & 31;
  const int w    = threadIdx.x >> 5;
  const int WNB = MLPD / 128; // 32
  const int mBlock = (blockIdx.x / WNB) * 128;
  const int nBlock = (blockIdx.x % WNB) * 128;
  v8f acc[2][4];
  gemm_block_128x128(A, Bt, DIMX, mBlock, nBlock, sA, sB, acc);
  const int mBaseW = mBlock + (w >> 1) * 32;
  const int nBaseW = nBlock + (w & 1) * 64;
  const int mo = (lane >> 4) << 3;
  const int no = lane & 15;
#pragma unroll
  for (int mt = 0; mt < 2; ++mt)
#pragma unroll
    for (int nt = 0; nt < 4; ++nt)
#pragma unroll
      for (int r = 0; r < 8; ++r) {
        const int row = mBaseW + 16 * mt + mo + r;
        const int col = nBaseW + 16 * nt + no;
        const float t = acc[mt][nt][r] + bias[col];
        const float gl = 0.5f * t * (1.f + erff(t * 0.70710678118654752f));
        outp[(size_t)row * MLPD + col] = (bf16_t)gl;
      }
}

// ---------------- flash attention ----------------
// One block = 8 waves = 8 query-tiles of the SAME (b,h); K/V tiles are staged
// once per 32-key step into LDS (async, triple buffered via rotating
// pointers) and shared by all waves. S^T = K x Q^T so each lane owns the
// scores of its own query.
__global__ __launch_bounds__(256) void k_flash_attn(
    const bf16_t* __restrict__ qb, const bf16_t* __restrict__ kbuf,
    const bf16_t* __restrict__ vtb, bf16_t* __restrict__ outb) {
  __shared__ __align__(16) bf16_t sK[3 * 32 * 64]; // [keys][d]
  __shared__ __align__(16) bf16_t sV[3 * 64 * 32]; // [d][keys]
  const int lane = threadIdx.x & 31;
  const int w    = threadIdx.x >> 5;
  const int bh = blockIdx.x >> 4;            // 16 blocks per (b,h)
  const int qt = (blockIdx.x & 15) * 8 + w;  // query tile 0..127
  const bf16_t* Q  = qb   + (size_t)bh * SEQ * DHEAD;
  const bf16_t* Kt = kbuf + (size_t)bh * SEQ * DHEAD;
  const bf16_t* Vt = vtb  + (size_t)bh * DHEAD * SEQ; // [d][n]
  const int qBase = qt * 16;

  // Q as B-fragment (n = query, K = d), loaded once: d 0..31 and 32..63
  const v16bf qf0 = load_fragB(Q, qBase, DHEAD, 0, lane);
  const v16bf qf1 = load_fragB(Q, qBase, DHEAD, 32, lane);

  v8f o[4];
#pragma unroll
  for (int t = 0; t < 4; ++t) o[t] = v8f_zero();
  float mrun = -1e30f, lrun = 0.f;

  // cooperative K/V tile staging: 256 threads, 2 x 16B chunks each
  auto stage_kv = [&](int kb0, bf16_t* dK, bf16_t* dV) {
    const int t = threadIdx.x;
    {
      const int row = t >> 3, ch = t & 7; // K: 32 rows x 8 chunks
      async_copy_b128(lds_off32(dK + row * DHEAD + ch * 8),
                      Kt + (size_t)(kb0 + row) * DHEAD + ch * 8);
    }
    {
      const int row = t >> 2, ch = t & 3; // V^T: 64 rows x 4 chunks
      async_copy_b128(lds_off32(dV + row * 32 + ch * 8),
                      Vt + (size_t)row * SEQ + kb0 + ch * 8);
    }
  };

  const int S = SEQ / 32; // 64 steps
  bf16_t* rdK = sK;           bf16_t* rdV = sV;
  bf16_t* wrK = sK + 2048;    bf16_t* wrV = sV + 2048;
  bf16_t* spK = sK + 4096;    bf16_t* spV = sV + 4096;
  stage_kv(0, rdK, rdV);
#pragma clang loop unroll(disable)
  for (int i = 0; i < S; ++i) {
    if (i + 1 < S) {
      stage_kv((i + 1) * 32, wrK, wrV);
      asm volatile("s_wait_asynccnt 0x2" ::: "memory");
    } else {
      asm volatile("s_wait_asynccnt 0x0" ::: "memory");
    }
    __syncthreads();

    // S^T tiles for keys 0..15 (s0) and 16..31 (s1) of this step
    v16bf kf = load_fragA(rdK, 0, DHEAD, 0, lane);
    v8f s0 = wmma_bf16f32(kf, qf0, v8f_zero());
    kf = load_fragA(rdK, 0, DHEAD, 32, lane);
    s0 = wmma_bf16f32(kf, qf1, s0);
    kf = load_fragA(rdK, 16, DHEAD, 0, lane);
    v8f s1 = wmma_bf16f32(kf, qf0, v8f_zero());
    kf = load_fragA(rdK, 16, DHEAD, 32, lane);
    s1 = wmma_bf16f32(kf, qf1, s1);

    float tmax = -1e30f;
#pragma unroll
    for (int r = 0; r < 8; ++r) {
      s0[r] *= 0.125f;  // dhead^-0.5
      s1[r] *= 0.125f;
      tmax = fmaxf(tmax, fmaxf(s0[r], s1[r]));
    }
    tmax = fmaxf(tmax, dswz<SWAPX16>(tmax)); // lane pair shares a query
    const float mnew  = fmaxf(mrun, tmax);
    const float alpha = __expf(mrun - mnew);

    float psum = 0.f;
    unsigned u0[4], u1[4];
#pragma unroll
    for (int j = 0; j < 4; ++j) {
      const float a0 = __expf(s0[2 * j] - mnew);
      const float a1 = __expf(s0[2 * j + 1] - mnew);
      const float b0 = __expf(s1[2 * j] - mnew);
      const float b1 = __expf(s1[2 * j + 1] - mnew);
      psum += a0 + a1 + b0 + b1;
      u0[j] = pack2bf(a0, a1);
      u1[j] = pack2bf(b0, b1);
    }
    lrun = lrun * alpha + psum;
    mrun = mnew;

    // Build P^T as B-fragment (K = 32 keys, N = 16 queries).
    // lane<16 needs keys {0..7 (own s0), 8..15 (partner s0)};
    // lane>=16 needs keys {16..23 (partner s1), 24..31 (own s1)}.
    union { v8u u; v16bf b; } pf;
    const bool lowHalf = (lane < 16);
#pragma unroll
    for (int j = 0; j < 4; ++j) {
      const unsigned sw0 = dswzu<SWAPX16>(u0[j]);
      const unsigned sw1 = dswzu<SWAPX16>(u1[j]);
      pf.u[j]     = lowHalf ? u0[j] : sw1;
      pf.u[j + 4] = lowHalf ? sw0 : u1[j];
    }

    // O^T += V^T x P^T   (4 tiles of 16 d-values)
#pragma unroll
    for (int t = 0; t < 4; ++t) {
#pragma unroll
      for (int r = 0; r < 8; ++r) o[t][r] *= alpha;
      const v16bf vf = load_fragA(rdV, t * 16, 32, 0, lane);
      o[t] = wmma_bf16f32(vf, pf.b, o[t]);
    }

    bf16_t* tK = rdK; rdK = wrK; wrK = spK; spK = tK; // rotate triple buffer
    bf16_t* tV = rdV; rdV = wrV; wrV = spV; spV = tV;
  }

  const float ltot = lrun + dswz<SWAPX16>(lrun);
  const float inv  = 1.f / ltot;
  const int b = bh >> 4, h = bh & 15;
  const int q = lane & 15;
  const int doff = (lane >> 4) << 3;
  bf16_t* orow =
      outb + ((size_t)b * SEQ + qBase + q) * DIMX + h * DHEAD + doff;
#pragma unroll
  for (int t = 0; t < 4; ++t)
#pragma unroll
    for (int r = 0; r < 8; ++r)
      orow[t * 16 + r] = (bf16_t)(o[t][r] * inv);
}

// ---------------- host-side orchestration ----------------

extern "C" void kernel_launch(void* const* d_in, const int* in_sizes, int n_in,
                              void* d_out, int out_size, void* d_ws,
                              size_t ws_size, hipStream_t stream) {
  (void)in_sizes; (void)n_in; (void)out_size; (void)ws_size;
  const float* x     = (const float*)d_in[0];
  const float* ln1_g = (const float*)d_in[1];
  const float* ln1_b = (const float*)d_in[2];
  const float* w_qkv = (const float*)d_in[3];
  const float* w_out = (const float*)d_in[4];
  const float* b_out = (const float*)d_in[5];
  const float* ln2_g = (const float*)d_in[6];
  const float* ln2_b = (const float*)d_in[7];
  const float* w1    = (const float*)d_in[8];
  const float* b1    = (const float*)d_in[9];
  const float* w2    = (const float*)d_in[10];
  const float* b2    = (const float*)d_in[11];
  float* out = (float*)d_out;

  char* ws = (char*)d_ws;
  size_t off = 0;
  auto alloc = [&](size_t bytes) -> void* {
    void* p = ws + off;
    off += (bytes + 255) & ~(size_t)255;
    return p;
  };
  bf16_t* hln   = (bf16_t*)alloc((size_t)ROWS * DIMX * 2);
  bf16_t* h2    = (bf16_t*)alloc((size_t)ROWS * DIMX * 2);
  bf16_t* wqkvT = (bf16_t*)alloc((size_t)3 * DIMX * DIMX * 2);
  bf16_t* woutT = (bf16_t*)alloc((size_t)DIMX * DIMX * 2);
  bf16_t* w1T   = (bf16_t*)alloc((size_t)DIMX * MLPD * 2);
  bf16_t* w2T   = (bf16_t*)alloc((size_t)DIMX * MLPD * 2);
  bf16_t* qbuf  = (bf16_t*)alloc((size_t)ROWS * DIMX * 2);
  bf16_t* kbuf  = (bf16_t*)alloc((size_t)ROWS * DIMX * 2);
  bf16_t* vtbuf = (bf16_t*)alloc((size_t)ROWS * DIMX * 2);
  bf16_t* aout  = (bf16_t*)alloc((size_t)ROWS * DIMX * 2);
  bf16_t* h1    = (bf16_t*)alloc((size_t)ROWS * MLPD * 2);
  float*  xattn = (float*)alloc((size_t)ROWS * DIMX * 4);

  // weight prep: fp32 [K][N] -> bf16 [N][K]
  k_transpose_cvt<<<(3 * DIMX * DIMX + 255) / 256, 256, 0, stream>>>(
      w_qkv, wqkvT, DIMX, 3 * DIMX);
  k_transpose_cvt<<<(DIMX * DIMX + 255) / 256, 256, 0, stream>>>(
      w_out, woutT, DIMX, DIMX);
  k_transpose_cvt<<<(DIMX * MLPD + 255) / 256, 256, 0, stream>>>(
      w1, w1T, DIMX, MLPD);
  k_transpose_cvt<<<(DIMX * MLPD + 255) / 256, 256, 0, stream>>>(
      w2, w2T, MLPD, DIMX);

  // attention half
  k_layernorm<<<ROWS, 256, 0, stream>>>(x, ln1_g, ln1_b, hln);
  k_gemm_qkv<<<(ROWS / 128) * ((3 * DIMX) / 128), 256, 0, stream>>>(
      hln, wqkvT, qbuf, kbuf, vtbuf);
  k_flash_attn<<<BATCH * HEADS * 16, 256, 0, stream>>>(
      qbuf, kbuf, vtbuf, aout);
  k_gemm_bias_res<<<(ROWS / 128) * (DIMX / 128), 256, 0, stream>>>(
      aout, woutT, b_out, x, xattn, DIMX, DIMX);

  // MLP half
  k_layernorm<<<ROWS, 256, 0, stream>>>(xattn, ln2_g, ln2_b, h2);
  k_gemm_mlp1<<<(ROWS / 128) * (MLPD / 128), 256, 0, stream>>>(
      h2, w1T, b1, h1);
  k_gemm_bias_res<<<(ROWS / 128) * (DIMX / 128), 256, 0, stream>>>(
      h1, w2T, b2, xattn, out, MLPD, DIMX);
}
